// GCN_67594195304512
// MI455X (gfx1250) — matrix operand order repaired and checked by reading, not compiled, not run on hardware
//
#include <hip/hip_runtime.h>
#include <hip/hip_bf16.h>
#include <math.h>

typedef __bf16 bf16_t;
typedef __bf16 v16bf __attribute__((ext_vector_type(16)));
typedef __bf16 v8bf  __attribute__((ext_vector_type(8)));
typedef float  v8f   __attribute__((ext_vector_type(8)));

#define GCN_F 128
#define GCN_H 1024
#define GCN_C 2

// ---- bf16 <-> f32 helpers (bit-exact RNE, no reliance on __bf16 arithmetic) ----
__device__ __forceinline__ bf16_t f2bf(float f) {
    unsigned u = __builtin_bit_cast(unsigned, f);
    unsigned r = u + 0x7fffu + ((u >> 16) & 1u);   // round-to-nearest-even
    unsigned short h = (unsigned short)(r >> 16);
    return __builtin_bit_cast(bf16_t, h);
}
__device__ __forceinline__ float bf2f(bf16_t b) {
    unsigned short h = __builtin_bit_cast(unsigned short, b);
    unsigned u = ((unsigned)h) << 16;
    return __builtin_bit_cast(float, u);
}

// ---------------- small utility kernels ----------------
__global__ void zero_f32(float* __restrict__ p, int n) {
    int i = blockIdx.x * blockDim.x + threadIdx.x;
    if (i < n) p[i] = 0.0f;
}

__global__ void count_deg(const int* __restrict__ dst, float* __restrict__ deg, int E) {
    int e = blockIdx.x * blockDim.x + threadIdx.x;
    if (e < E) atomicAdd(&deg[dst[e]], 1.0f);
}

__global__ void compute_dinv(const float* __restrict__ deg, float* __restrict__ dinv, int n) {
    int i = blockIdx.x * blockDim.x + threadIdx.x;
    if (i < n) dinv[i] = rsqrtf(deg[i] + 1.0f);   // +1 = self loop
}

// xprop[i][:] = dinv[i]^2 * x[i][:]   (self-loop contribution, also zero-inits)
__global__ void init_xprop(const float* __restrict__ x, const float* __restrict__ dinv,
                           float* __restrict__ xprop, int n) {
    int i = blockIdx.x * blockDim.x + threadIdx.x;
    if (i < n * GCN_F) {
        int node = i >> 7;
        float d = dinv[node];
        xprop[i] = d * d * x[i];
    }
}

// one wave per edge; lane handles 4 consecutive features (128 = 32 lanes * 4)
__global__ __launch_bounds__(256) void scatter_x(const int* __restrict__ src,
                                                 const int* __restrict__ dst,
                                                 const float* __restrict__ dinv,
                                                 const float* __restrict__ x,
                                                 float* __restrict__ xprop, int E) {
    int e = blockIdx.x * 8 + (threadIdx.x >> 5);
    if (e >= E) return;
    int lane = threadIdx.x & 31;
    int s = src[e], d = dst[e];
    float nrm = dinv[s] * dinv[d];
    float4 v = *(const float4*)(x + (size_t)s * GCN_F + lane * 4);
    float* o = xprop + (size_t)d * GCN_F + lane * 4;
    atomicAdd(o + 0, nrm * v.x);
    atomicAdd(o + 1, nrm * v.y);
    atomicAdd(o + 2, nrm * v.z);
    atomicAdd(o + 3, nrm * v.w);
}

// ---------------- GEMM1: h = relu(xprop @ W1 + b1), bf16 WMMA, f32 accum ----------------
// block tile 64(M) x 128(N), full K=128 in LDS; 8 waves -> each 16x64 (4 wmma tiles x 4 K-steps)
__global__ __launch_bounds__(256) void gemm1_wmma(const float* __restrict__ xprop,
                                                  const float* __restrict__ W1,
                                                  const float* __restrict__ b1,
                                                  bf16_t* __restrict__ h, int n_nodes) {
    __shared__ __align__(16) bf16_t As[64][136];    // [m][k], pad to keep 16B rows aligned
    __shared__ __align__(16) bf16_t Bs[128][136];   // [n][k] (W1 transposed)

    const int tid = threadIdx.x;
    const int bm = blockIdx.x * 64;
    const int bn = blockIdx.y * 128;

    // stage A tile (f32 -> bf16), zero-fill OOB rows
    for (int i = tid; i < (64 * 128) / 4; i += 256) {
        int idx = i * 4;
        int row = idx >> 7;
        int k   = idx & 127;
        int gr  = bm + row;
        float4 v = make_float4(0.f, 0.f, 0.f, 0.f);
        if (gr < n_nodes) v = *(const float4*)(xprop + (size_t)gr * GCN_F + k);
        As[row][k + 0] = f2bf(v.x);
        As[row][k + 1] = f2bf(v.y);
        As[row][k + 2] = f2bf(v.z);
        As[row][k + 3] = f2bf(v.w);
    }
    // stage B slab transposed: Bs[n][k] = W1[k][bn+n]  (coalesced along n)
    for (int i = tid; i < 128 * 128; i += 256) {
        int k = i >> 7;
        int n = i & 127;
        Bs[n][k] = f2bf(W1[(size_t)k * GCN_H + bn + n]);
    }
    __syncthreads();

    const int lane = tid & 31;
    const int wave = tid >> 5;
    const int wm = wave & 3;      // 4 wave-rows of 16
    const int wn = wave >> 2;     // 2 wave-cols of 64
    const int lr = lane & 15;
    const int hi = lane >> 4;

    v8f acc[4] = {v8f{}, v8f{}, v8f{}, v8f{}};

    #pragma unroll
    for (int kk = 0; kk < 128; kk += 32) {
        // A fragment (ISA 16-bit A 16x32): lane<16 -> K kk+0..7 & kk+16..23; lane>=16 -> +8
        const bf16_t* ap = &As[wm * 16 + lr][kk + hi * 8];
        v8bf a0 = *(const v8bf*)ap;
        v8bf a1 = *(const v8bf*)(ap + 16);
        v16bf a = __builtin_shufflevector(a0, a1, 0,1,2,3,4,5,6,7,8,9,10,11,12,13,14,15);
        #pragma unroll
        for (int j = 0; j < 4; ++j) {
            // B fragment: col = lane%16, lane<16 -> K kk+0..7 & kk+16..23; lane>=16 -> +8
            const bf16_t* bp = &Bs[wn * 64 + j * 16 + lr][kk + hi * 8];
            v8bf b0 = *(const v8bf*)bp;
            v8bf b1 = *(const v8bf*)(bp + 16);
            v16bf b = __builtin_shufflevector(b0, b1, 0,1,2,3,4,5,6,7,8,9,10,11,12,13,14,15);
            acc[j] = __builtin_amdgcn_wmma_f32_16x16x32_bf16(
                false, a, false, b, (short)0, acc[j], false, false);
        }
    }

    // epilogue: C/D layout -> lane<16: M=r, lane>=16: M=r+8; N=lane%16
    #pragma unroll
    for (int j = 0; j < 4; ++j) {
        int col = bn + wn * 64 + j * 16 + lr;
        float bias = b1[col];
        #pragma unroll
        for (int r = 0; r < 8; ++r) {
            int row = bm + wm * 16 + r + hi * 8;
            if (row < n_nodes) {
                float v = acc[j][r] + bias;
                v = v > 0.f ? v : 0.f;                       // ReLU
                h[(size_t)row * GCN_H + col] = f2bf(v);
            }
        }
    }
}

// ---------------- GEMM2: z2 = h @ W2  (wave per node, K=1024, C=2) ----------------
__global__ __launch_bounds__(256) void gemm2_dot(const bf16_t* __restrict__ h,
                                                 const float* __restrict__ W2,
                                                 float* __restrict__ z2, int n_nodes) {
    int node = blockIdx.x * 8 + (threadIdx.x >> 5);
    if (node >= n_nodes) return;
    int lane = threadIdx.x & 31;
    const bf16_t* hr = h + (size_t)node * GCN_H + lane * 32;
    float a0 = 0.f, a1 = 0.f;
    #pragma unroll
    for (int c = 0; c < 4; ++c) {
        v8bf hv = *(const v8bf*)(hr + c * 8);     // 16B vector load
        int kb = lane * 32 + c * 8;
        #pragma unroll
        for (int i = 0; i < 8; ++i) {
            float xv = bf2f(hv[i]);
            a0 += xv * W2[(kb + i) * 2 + 0];
            a1 += xv * W2[(kb + i) * 2 + 1];
        }
    }
    #pragma unroll
    for (int off = 16; off > 0; off >>= 1) {
        a0 += __shfl_xor(a0, off, 32);
        a1 += __shfl_xor(a1, off, 32);
    }
    if (lane == 0) {
        z2[(size_t)node * 2 + 0] = a0;
        z2[(size_t)node * 2 + 1] = a1;
    }
}

// z[i][:] = dinv[i]^2 * z2[i][:]   (self loop, also zero-inits accumulator)
__global__ void init_z(const float* __restrict__ z2, const float* __restrict__ dinv,
                       float* __restrict__ z, int n) {
    int i = blockIdx.x * blockDim.x + threadIdx.x;
    if (i < n) {
        float d2 = dinv[i] * dinv[i];
        z[i * 2 + 0] = d2 * z2[i * 2 + 0];
        z[i * 2 + 1] = d2 * z2[i * 2 + 1];
    }
}

__global__ void scatter_z(const int* __restrict__ src, const int* __restrict__ dst,
                          const float* __restrict__ dinv, const float* __restrict__ z2,
                          float* __restrict__ z, int E) {
    int e = blockIdx.x * blockDim.x + threadIdx.x;
    if (e < E) {
        int s = src[e], d = dst[e];
        float nrm = dinv[s] * dinv[d];
        atomicAdd(&z[(size_t)d * 2 + 0], nrm * z2[(size_t)s * 2 + 0]);
        atomicAdd(&z[(size_t)d * 2 + 1], nrm * z2[(size_t)s * 2 + 1]);
    }
}

__global__ void finalize_softmax(const float* __restrict__ z, const float* __restrict__ b2,
                                 float* __restrict__ out, int n) {
    int i = blockIdx.x * blockDim.x + threadIdx.x;
    if (i < n) {
        float a = z[i * 2 + 0] + b2[0];
        float b = z[i * 2 + 1] + b2[1];
        float m = fmaxf(a, b);
        float e0 = expf(a - m), e1 = expf(b - m);
        float inv = 1.0f / (e0 + e1);
        out[i * 2 + 0] = e0 * inv;
        out[i * 2 + 1] = e1 * inv;
    }
}

// ---------------- launch ----------------
extern "C" void kernel_launch(void* const* d_in, const int* in_sizes, int n_in,
                              void* d_out, int out_size, void* d_ws, size_t ws_size,
                              hipStream_t stream) {
    const float* x  = (const float*)d_in[0];
    const int*   ei = (const int*)d_in[1];
    const float* W1 = (const float*)d_in[2];
    const float* b1 = (const float*)d_in[3];
    const float* W2 = (const float*)d_in[4];
    const float* b2 = (const float*)d_in[5];
    float* out = (float*)d_out;

    const int N = in_sizes[0] / GCN_F;
    const int E = in_sizes[1] / 2;
    const int* src = ei;        // edge_index row 0
    const int* dst = ei + E;    // edge_index row 1

    // carve workspace (256B aligned regions)
    char* w = (char*)d_ws;
    auto alloc = [&](size_t bytes) {
        char* p = w;
        w += (bytes + 255) & ~(size_t)255;
        return p;
    };
    float*  deg   = (float*) alloc((size_t)N * 4);
    float*  dinv  = (float*) alloc((size_t)N * 4);
    float*  xprop = (float*) alloc((size_t)N * GCN_F * 4);
    bf16_t* hbuf  = (bf16_t*)alloc((size_t)N * GCN_H * 2);
    float*  z2    = (float*) alloc((size_t)N * 2 * 4);
    float*  zacc  = (float*) alloc((size_t)N * 2 * 4);

    zero_f32<<<(N + 255) / 256, 256, 0, stream>>>(deg, N);
    count_deg<<<(E + 255) / 256, 256, 0, stream>>>(dst, deg, E);
    compute_dinv<<<(N + 255) / 256, 256, 0, stream>>>(deg, dinv, N);
    init_xprop<<<((N * GCN_F) + 255) / 256, 256, 0, stream>>>(x, dinv, xprop, N);
    scatter_x<<<(E + 7) / 8, 256, 0, stream>>>(src, dst, dinv, x, xprop, E);

    dim3 g1((N + 63) / 64, GCN_H / 128);
    gemm1_wmma<<<g1, 256, 0, stream>>>(xprop, W1, b1, hbuf, N);

    gemm2_dot<<<(N + 7) / 8, 256, 0, stream>>>(hbuf, W2, z2, N);
    init_z<<<(N + 255) / 256, 256, 0, stream>>>(z2, dinv, zacc, N);
    scatter_z<<<(E + 255) / 256, 256, 0, stream>>>(src, dst, dinv, z2, zacc, E);
    finalize_softmax<<<(N + 255) / 256, 256, 0, stream>>>(zacc, b2, out, N);
}
